// GlobalHeadProbEncoder_43662637531769
// MI455X (gfx1250) — compile-verified, matrix operand order checked by reading, not run on hardware
//
#include <hip/hip_runtime.h>
#include <hip/hip_bf16.h>

// ---------------- problem constants ----------------
#define BB 16
#define NN 256
#define DD 128
#define HH 8
#define GG 64
#define NITER 4
#define NEGV -1000000000.0f

typedef __attribute__((ext_vector_type(16))) __bf16 v16bf;
typedef __attribute__((ext_vector_type(8)))  float  v8f;
typedef unsigned short u16;

// float -> bf16 (round-to-nearest-even)
__device__ __forceinline__ u16 f2bf(float x) {
    unsigned u = __float_as_uint(x);
    unsigned r = u + 0x7FFFu + ((u >> 16) & 1u);
    return (u16)(r >> 16);
}

// Load one 16x32 bf16 WMMA operand (A layout; B operand = A layout of B^T).
// Row-major source, ld in elements. Lane L<16 -> row L, K chunks [k0, k0+8) and
// [k0+16, k0+24); lane L+16 -> row L, chunks [k0+8,k0+16) and [k0+24,k0+32).
// Works for both global and LDS-resident sources (compiler picks vmem/ds path).
__device__ __forceinline__ v16bf load_op(const u16* base, int ld, int row0, int k0) {
    int lane = threadIdx.x & 31;
    int r = lane & 15, half = lane >> 4;
    const u16* p = base + (size_t)(row0 + r) * ld + k0 + 8 * half;
    union { uint4 q[2]; v16bf v; } u;
    u.q[0] = *(const uint4*)(p);
    u.q[1] = *(const uint4*)(p + 16);
    return u.v;
}

__device__ __forceinline__ v8f wmma_bf16(v16bf a, v16bf b, v8f c) {
    return __builtin_amdgcn_wmma_f32_16x16x32_bf16(false, a, false, b, (short)0, c, false, false);
}

// ---------------- K0: pack weights to bf16 slices ----------------
// Tn[k][h][a][c] = ternary[k,a,c,h]      (B^T for right GEMM)
// Tt[k][h][c][a] = ternary[k,a,c,h]      (B^T for left GEMM)
// Gga[h][g][a]   = global_[g,a,h]        (B^T for Gs GEMM)
// Gag[h][a][g]   = global_[g,a,h]        (B^T for MG GEMM)
__global__ __launch_bounds__(256) void k_pack(const float* __restrict__ ternary,
                                              const float* __restrict__ glb,
                                              u16* __restrict__ Tn, u16* __restrict__ Tt,
                                              u16* __restrict__ Gga, u16* __restrict__ Gag) {
    int id = blockIdx.x * 256 + threadIdx.x;
    if (id < 2 * HH * DD * DD) {
        int c = id & 127, a = (id >> 7) & 127, h = (id >> 14) & 7, k = id >> 17;
        u16 v = f2bf(ternary[(((size_t)k * DD + a) * DD + c) * HH + h]);
        Tn[(((size_t)(k * HH + h)) * DD + a) * DD + c] = v;
        Tt[(((size_t)(k * HH + h)) * DD + c) * DD + a] = v;
    } else {
        int id2 = id - 2 * HH * DD * DD;   // < HH*DD*GG
        int a = id2 & 127, g = (id2 >> 7) & 63, h = id2 >> 13;
        u16 v = f2bf(glb[((size_t)g * DD + a) * HH + h]);
        Gga[((size_t)h * GG + g) * DD + a] = v;
        Gag[((size_t)h * DD + a) * GG + g] = v;
    }
}

// ---------------- K1: row softmax over D=128, f32 -> bf16 ----------------
__global__ __launch_bounds__(256) void k_softmax128(const float* __restrict__ src,
                                                    u16* __restrict__ qzb) {
    int wid = threadIdx.x >> 5, lane = threadIdx.x & 31;
    int row = blockIdx.x * 8 + wid;            // BB*NN = 4096 rows
    const float4 v = *(const float4*)(src + (size_t)row * DD + lane * 4);
    float m = fmaxf(fmaxf(v.x, v.y), fmaxf(v.z, v.w));
    for (int off = 16; off; off >>= 1) m = fmaxf(m, __shfl_xor(m, off, 32));
    float ex = __expf(v.x - m), ey = __expf(v.y - m);
    float ez = __expf(v.z - m), ew = __expf(v.w - m);
    float s = ex + ey + ez + ew;
    for (int off = 16; off; off >>= 1) s += __shfl_xor(s, off, 32);
    float inv = 1.0f / s;
    ushort4 o;
    o.x = f2bf(ex * inv); o.y = f2bf(ey * inv); o.z = f2bf(ez * inv); o.w = f2bf(ew * inv);
    *(ushort4*)(qzb + (size_t)row * DD + lane * 4) = o;
}

// ---------------- K2: left/leftT/rightT GEMMs per (k,b,h) ----------------
// left[k,b,h][i][c]  = sum_a qz[b][i][a] * Tt[k,h][c][a]^T     (256x128)
// right[k,b,h][j][a] = sum_c qz[b][j][c] * Tn[k,h][a][c]^T  -> stored transposed only
// Both weight slices (16x reuse across row-tiles) are staged in LDS once per block.
__global__ __launch_bounds__(256) void k_leftright(const u16* __restrict__ qzb,
                                                   const u16* __restrict__ Tn,
                                                   const u16* __restrict__ Tt,
                                                   u16* __restrict__ left,
                                                   u16* __restrict__ leftT,
                                                   u16* __restrict__ rightT) {
    __shared__ u16 sT[2 * DD * DD];            // [0..): Tt slice, [DD*DD..): Tn slice (64 KB)
    int combo = blockIdx.x;                    // 2*16*8 = 256
    int k = combo >> 7, b = (combo >> 3) & 15, h = combo & 7;
    int wid = threadIdx.x >> 5, lane = threadIdx.x & 31;
    int n = lane & 15, mb = (lane >> 4) * 8;
    const u16* qz = qzb + (size_t)b * NN * DD;
    size_t khs = (size_t)(k * HH + h) * DD * DD;

    // cooperative stage of both slices: 2 x 2048 uint4
    {
        const uint4* s0 = (const uint4*)(Tt + khs);
        const uint4* s1 = (const uint4*)(Tn + khs);
        uint4* d0 = (uint4*)sT;
        uint4* d1 = (uint4*)(sT + DD * DD);
        for (int t = threadIdx.x; t < DD * DD / 8; t += 256) {
            d0[t] = s0[t];
            d1[t] = s1[t];
        }
    }
    __syncthreads();

    size_t kbh = ((size_t)k * BB + b) * HH + h;
    u16* Lp  = left   + kbh * NN * DD;
    u16* LTp = leftT  + kbh * DD * NN;
    u16* RTp = rightT + kbh * DD * NN;

    for (int t = wid; t < 256; t += 8) {
        bool isLeft = t < 128;
        int tt = isLeft ? t : t - 128;
        int r0 = (tt >> 3) * 16, c0 = (tt & 7) * 16;
        const u16* Bop = isLeft ? sT : (sT + DD * DD);
        v8f acc = {};
        for (int kc = 0; kc < DD; kc += 32) {
            v16bf a = load_op(qz, DD, r0, kc);
            v16bf bm = load_op(Bop, DD, c0, kc);    // ds path
            acc = wmma_bf16(a, bm, acc);
        }
        if (isLeft) {
            for (int e = 0; e < 8; ++e) {
                int i = r0 + mb + e, c = c0 + n;
                u16 v = f2bf(acc[e]);
                Lp[(size_t)i * DD + c]  = v;
                LTp[(size_t)c * NN + i] = v;       // transposed copy for M2 B-operand
            }
        } else {
            for (int e = 0; e < 8; ++e) {
                int j = r0 + mb + e, a2 = c0 + n;
                RTp[(size_t)a2 * NN + j] = f2bf(acc[e]);  // transposed for M1 B-operand
            }
        }
    }
}

// ---------------- K3: fused F/Gs + bias + softmax(320) + masked bf16 scatter ----
__global__ __launch_bounds__(128) void k_fused_s(const u16* __restrict__ qzb,
                                                 const u16* __restrict__ left,
                                                 const u16* __restrict__ Gga,
                                                 const int* __restrict__ mask,
                                                 u16* __restrict__ qhd0, u16* __restrict__ qhd1,
                                                 u16* __restrict__ qhd0T, u16* __restrict__ qhd1T,
                                                 u16* __restrict__ qhglb) {
    __shared__ float Sm[16 * 320];
    int blk = blockIdx.x;                      // 16*8*16 = 2048
    int it = blk & 15, h = (blk >> 4) & 7, b = blk >> 7;
    int wid = threadIdx.x >> 5, lane = threadIdx.x & 31;
    int n = lane & 15, mb = (lane >> 4) * 8;
    int i0 = it * 16;
    const u16* qz = qzb + (size_t)b * NN * DD;
    const int* mrow = mask + (size_t)b * NN;

    for (int t = wid; t < 20; t += 4) {
        if (t < 16) {
            int j0 = t * 16;
            if (t != it) {
                int kk = (t > it) ? 0 : 1;
                const u16* A = left + (((size_t)kk * BB + b) * HH + h) * NN * DD;
                v8f acc = {};
                for (int kc = 0; kc < DD; kc += 32) {
                    v16bf va = load_op(A, DD, i0, kc);
                    v16bf vb = load_op(qz, DD, j0, kc);
                    acc = wmma_bf16(va, vb, acc);
                }
                for (int e = 0; e < 8; ++e) {
                    int il = mb + e, i = i0 + il, j = j0 + n;
                    float v = acc[e];
                    if (!(mrow[i] && mrow[j])) v += NEGV;
                    Sm[il * 320 + j] = v;
                }
            } else {  // diagonal tile: both k's, per-element triangular select
                const u16* A0 = left + (((size_t)0 * BB + b) * HH + h) * NN * DD;
                const u16* A1 = left + (((size_t)1 * BB + b) * HH + h) * NN * DD;
                v8f a0 = {}, a1 = {};
                for (int kc = 0; kc < DD; kc += 32) {
                    v16bf vb = load_op(qz, DD, j0, kc);
                    a0 = wmma_bf16(load_op(A0, DD, i0, kc), vb, a0);
                    a1 = wmma_bf16(load_op(A1, DD, i0, kc), vb, a1);
                }
                for (int e = 0; e < 8; ++e) {
                    int il = mb + e, i = i0 + il, j = j0 + n;
                    float v = (n > il) ? a0[e] : ((n < il) ? a1[e] : 0.0f);
                    if (!(mrow[i] && mrow[j])) v += NEGV;
                    Sm[il * 320 + j] = v;
                }
            }
        } else {   // Gs tiles: cols [256,320)
            int g0 = (t - 16) * 16;
            const u16* Bop = Gga + (size_t)h * GG * DD;
            v8f acc = {};
            for (int kc = 0; kc < DD; kc += 32) {
                v16bf va = load_op(qz, DD, i0, kc);
                v16bf vb = load_op(Bop, DD, g0, kc);
                acc = wmma_bf16(va, vb, acc);
            }
            for (int e = 0; e < 8; ++e)
                Sm[(mb + e) * 320 + 256 + g0 + n] = acc[e];
        }
    }
    __syncthreads();

    // row softmax over 320 cols; one wave per row
    for (int row = wid; row < 16; row += 4) {
        int ig = i0 + row;
        float vals[10], mx = -3.4e38f;
        for (int t2 = 0; t2 < 10; ++t2) {
            vals[t2] = Sm[row * 320 + lane + 32 * t2];
            mx = fmaxf(mx, vals[t2]);
        }
        for (int off = 16; off; off >>= 1) mx = fmaxf(mx, __shfl_xor(mx, off, 32));
        float s = 0.0f;
        for (int t2 = 0; t2 < 10; ++t2) { vals[t2] = __expf(vals[t2] - mx); s += vals[t2]; }
        for (int off = 16; off; off >>= 1) s += __shfl_xor(s, off, 32);
        float inv = 1.0f / s;
        size_t rb = (size_t)(b * HH + h) * NN + ig;
        for (int t2 = 0; t2 < 10; ++t2) {
            int col = lane + 32 * t2;
            float p = vals[t2] * inv;
            if (col < NN) {
                int j = col;
                float up = (j > ig) ? p : 0.0f;
                float lo = (j < ig) ? p : 0.0f;
                size_t ij = rb * NN + j;
                size_t ji = ((size_t)(b * HH + h) * NN + j) * NN + ig;
                qhd0[ij]  = f2bf(up); qhd1[ij]  = f2bf(lo);
                qhd0T[ji] = f2bf(up); qhd1T[ji] = f2bf(lo);
            } else {
                qhglb[rb * GG + (col - NN)] = f2bf(p);
            }
        }
    }
}

// ---------------- K4: Q_Z = x + M1 + M2 + MG (fused, 272 WMMAs / tile) ------
// Block = (b, n-tile); 8 waves = 8 d-tiles. A-panels (16x256 bf16, identical for
// all 8 waves) are staged into double-buffered LDS -> 8x less L2 operand traffic.
__global__ __launch_bounds__(256) void k_out(const float* __restrict__ x,
                                             const u16* __restrict__ qhd0, const u16* __restrict__ qhd1,
                                             const u16* __restrict__ qhd0T, const u16* __restrict__ qhd1T,
                                             const u16* __restrict__ rightT, const u16* __restrict__ leftT,
                                             const u16* __restrict__ qhglb, const u16* __restrict__ Gag,
                                             float* __restrict__ out) {
    __shared__ u16 As[2][16 * NN];             // 2 x 8 KB ping-pong A panels
    int wid = threadIdx.x >> 5, lane = threadIdx.x & 31;
    int b = blockIdx.x >> 4, nt = blockIdx.x & 15;   // 256 blocks
    int n0 = nt * 16, d0 = wid * 16;
    int nl = lane & 15, mb = (lane >> 4) * 8;
    int pbuf = 0;

    v8f acc = {};
    for (int k = 0; k < 2; ++k) {
        const u16* qA  = k ? qhd1  : qhd0;
        const u16* qAT = k ? qhd1T : qhd0T;
        for (int h = 0; h < HH; ++h) {
            size_t bh  = (size_t)(b * HH + h) * NN;
            size_t kbh = (((size_t)k * BB + b) * HH + h) * DD;
            const u16* B1 = rightT + kbh * NN;   // right^T rows a, K=j
            const u16* B2 = leftT  + kbh * NN;   // left^T  rows c, K=i
            // ---- M1: A = qhd_k rows n0, K=j ----
            {
                const u16* src = qA + (bh + n0) * NN;
                int row = threadIdx.x >> 4, seg = (threadIdx.x & 15) * 16;
                uint4* d = (uint4*)(&As[pbuf][row * NN + seg]);
                const uint4* s = (const uint4*)(src + (size_t)row * NN + seg);
                d[0] = s[0]; d[1] = s[1];
            }
            __syncthreads();
            __builtin_prefetch(B1 + (size_t)d0 * NN, 0, 1);
            for (int kc = 0; kc < NN; kc += 32)
                acc = wmma_bf16(load_op(As[pbuf], NN, 0, kc), load_op(B1, NN, d0, kc), acc);
            pbuf ^= 1;
            // ---- M2: A = qhd_k^T rows n0, K=i ----
            {
                const u16* src = qAT + (bh + n0) * NN;
                int row = threadIdx.x >> 4, seg = (threadIdx.x & 15) * 16;
                uint4* d = (uint4*)(&As[pbuf][row * NN + seg]);
                const uint4* s = (const uint4*)(src + (size_t)row * NN + seg);
                d[0] = s[0]; d[1] = s[1];
            }
            __syncthreads();
            __builtin_prefetch(B2 + (size_t)d0 * NN, 0, 1);
            for (int kc = 0; kc < NN; kc += 32)
                acc = wmma_bf16(load_op(As[pbuf], NN, 0, kc), load_op(B2, NN, d0, kc), acc);
            pbuf ^= 1;
        }
    }
    for (int h = 0; h < HH; ++h) {             // MG: A = qh_glb rows n0, K=g (16x64 panel)
        const u16* src = qhglb + ((size_t)(b * HH + h) * NN + n0) * GG;
        if (threadIdx.x < 64) {
            int row = threadIdx.x >> 2, seg = (threadIdx.x & 3) * 16;
            uint4* d = (uint4*)(&As[pbuf][row * GG + seg]);
            const uint4* s = (const uint4*)(src + (size_t)row * GG + seg);
            d[0] = s[0]; d[1] = s[1];
        }
        __syncthreads();
        const u16* Bop = Gag + (size_t)h * DD * GG;
        for (int kc = 0; kc < GG; kc += 32)
            acc = wmma_bf16(load_op(As[pbuf], GG, 0, kc), load_op(Bop, GG, d0, kc), acc);
        pbuf ^= 1;
    }
    for (int e = 0; e < 8; ++e) {
        size_t idx = ((size_t)b * NN + n0 + mb + e) * DD + d0 + nl;
        out[idx] = x[idx] + acc[e];
    }
}

// ---------------- host launcher ----------------
extern "C" void kernel_launch(void* const* d_in, const int* in_sizes, int n_in,
                              void* d_out, int out_size, void* d_ws, size_t ws_size,
                              hipStream_t stream) {
    const float* x       = (const float*)d_in[0];   // (16,256,128) f32
    const int*   mask    = (const int*)  d_in[1];   // (16,256) i32
    const float* ternary = (const float*)d_in[2];   // (2,128,128,8) f32
    const float* glb     = (const float*)d_in[3];   // (64,128,8) f32
    float* out = (float*)d_out;

    char* w = (char*)d_ws;
    size_t off = 0;
    auto alloc = [&](size_t bytes) -> void* {
        void* p = w + off;
        off += (bytes + 255) & ~(size_t)255;
        return p;
    };
    float* QZ  = (float*)alloc((size_t)BB * NN * DD * 4);
    u16* Tn    = (u16*)alloc((size_t)2 * HH * DD * DD * 2);
    u16* Tt    = (u16*)alloc((size_t)2 * HH * DD * DD * 2);
    u16* Gga   = (u16*)alloc((size_t)HH * GG * DD * 2);
    u16* Gag   = (u16*)alloc((size_t)HH * DD * GG * 2);
    u16* qzb   = (u16*)alloc((size_t)BB * NN * DD * 2);
    u16* left  = (u16*)alloc((size_t)2 * BB * HH * NN * DD * 2);
    u16* leftT = (u16*)alloc((size_t)2 * BB * HH * DD * NN * 2);
    u16* rightT= (u16*)alloc((size_t)2 * BB * HH * DD * NN * 2);
    u16* qhd0  = (u16*)alloc((size_t)BB * HH * NN * NN * 2);
    u16* qhd1  = (u16*)alloc((size_t)BB * HH * NN * NN * 2);
    u16* qhd0T = (u16*)alloc((size_t)BB * HH * NN * NN * 2);
    u16* qhd1T = (u16*)alloc((size_t)BB * HH * NN * NN * 2);
    u16* qhglb = (u16*)alloc((size_t)BB * HH * NN * GG * 2);

    k_pack<<<1280, 256, 0, stream>>>(ternary, glb, Tn, Tt, Gga, Gag);

    for (int it = 0; it < NITER; ++it) {
        const float* src = it ? (const float*)QZ : x;
        k_softmax128<<<512, 256, 0, stream>>>(src, qzb);
        k_leftright<<<256, 256, 0, stream>>>(qzb, Tn, Tt, left, leftT, rightT);
        k_fused_s<<<2048, 128, 0, stream>>>(qzb, left, Gga, mask,
                                            qhd0, qhd1, qhd0T, qhd1T, qhglb);
        k_out<<<256, 256, 0, stream>>>(x, qhd0, qhd1, qhd0T, qhd1T,
                                       rightT, leftT, qhglb, Gag,
                                       (it < NITER - 1) ? QZ : out);
    }
}
